// HGTLayer_68109591380574
// MI455X (gfx1250) — matrix operand order repaired and checked by reading, not compiled, not run on hardware
//
#include <hip/hip_runtime.h>
#include <hip/hip_bf16.h>
#include <stdint.h>

// HGT layer for MI455X (gfx1250).
// Dense math: v_wmma_f32_16x16x32_bf16 with LDS-staged B tiles.
// Sparse softmax/aggregation: CSR build (~800K light atomics) + per-segment
// wave aggregation with zero heavy atomics. Workspace: ~225 MB.

#define NN 50000
#define EE 200000
#define RR 2
#define DD 256
#define HH 8
#define DKK 32
#define MTILES (NN / 16)        // 3125, exact
#define NPAIRS ((MTILES + 1) / 2) // 1563
#define RN (RR * NN)            // 100000 segments

typedef __attribute__((ext_vector_type(16))) __bf16 v16bf;
typedef __attribute__((ext_vector_type(8)))  float  v8f;
typedef unsigned short ushort_t;
typedef unsigned int   uint_t;

union ABFrag { uint4 q[2]; v16bf v; };

__device__ __forceinline__ ushort_t f2bf(float f) {
    uint_t u = __float_as_uint(f);
    u += 0x7FFFu + ((u >> 16) & 1u);   // round-to-nearest-even
    return (ushort_t)(u >> 16);
}
__device__ __forceinline__ float bflo(uint_t u) { return __uint_as_float(u << 16); }
__device__ __forceinline__ float bfhi(uint_t u) { return __uint_as_float(u & 0xFFFF0000u); }

// ---------------- utility kernels ----------------

__global__ void zero16_kernel(uint4* p, int n16) {
    int i = blockIdx.x * blockDim.x + threadIdx.x;
    if (i < n16) p[i] = make_uint4(0u, 0u, 0u, 0u);
}

__global__ void conv_x_kernel(const float4* __restrict__ x, ushort_t* __restrict__ xb, int n4) {
    int i = blockIdx.x * blockDim.x + threadIdx.x;
    if (i >= n4) return;
    float4 f = x[i];
    uint2 o;
    o.x = (uint_t)f2bf(f.x) | ((uint_t)f2bf(f.y) << 16);
    o.y = (uint_t)f2bf(f.z) | ((uint_t)f2bf(f.w) << 16);
    ((uint2*)xb)[i] = o;
}

// Wk, Wv -> bf16, transposed to column-major [col][row] for B-fragments.
__global__ void prep_wkv_kernel(const float* __restrict__ Wk, const float* __restrict__ Wv,
                                ushort_t* __restrict__ wk_t, ushort_t* __restrict__ wv_t) {
    int idx = blockIdx.x * blockDim.x + threadIdx.x;
    if (idx >= 2 * DD * DD) return;
    int which = idx >> 16;
    int c = (idx >> 8) & 255;
    int k = idx & 255;
    const float* W = which ? Wv : Wk;
    ushort_t* O = which ? wv_t : wk_t;
    O[c * DD + k] = f2bf(W[k * DD + c]);
}

// Combined query weight: Wq_r[k, h*32+d] = sum_f Wq[k,h*32+f]*A[r,h,d,f]*pri[r,h]/sqrt(DK)
__global__ void prep_wq_kernel(const float* __restrict__ Wq, const float* __restrict__ bq,
                               const float* __restrict__ rel_att, const float* __restrict__ rel_pri,
                               ushort_t* __restrict__ wq_t, float* __restrict__ bq_c) {
    int idx = blockIdx.x * blockDim.x + threadIdx.x;
    if (idx >= RR * DD * DD) return;
    int r = idx >> 16;
    int c = (idx >> 8) & 255;   // output column h*32+d
    int k = idx & 255;          // input row
    int h = c >> 5, d = c & 31;
    float scale = rel_pri[r * HH + h] * 0.17677669529663687f; // 1/sqrt(32)
    const float* A = rel_att + (((size_t)(r * HH + h) * DKK) + d) * DKK;
    const float* wrow = Wq + (size_t)k * DD + (h << 5);
    float sum = 0.f;
#pragma unroll
    for (int f = 0; f < DKK; ++f) sum += wrow[f] * A[f];
    wq_t[(size_t)r * DD * DD + (size_t)c * DD + k] = f2bf(sum * scale);
    if (k == 0) {
        const float* brow = bq + (h << 5);
        float bs = 0.f;
#pragma unroll
        for (int f = 0; f < DKK; ++f) bs += brow[f] * A[f];
        bq_c[r * DD + c] = bs * scale;
    }
}

// Combined output weight: Wfin[r*256+h*32+d1, j] = 0.5*sum_d2 M[r,h,d1,d2]*Wa[h*32+d2, j]
__global__ void prep_wfin_kernel(const float* __restrict__ Wa, const float* __restrict__ rel_msg,
                                 ushort_t* __restrict__ wfin_t) {
    int idx = blockIdx.x * blockDim.x + threadIdx.x;
    if (idx >= 2 * DD * DD) return;
    int j = idx & 255;
    int kk = idx >> 8;           // 0..511
    int r = kk >> 8;
    int c1 = kk & 255;
    int h = c1 >> 5, d1 = c1 & 31;
    const float* M = rel_msg + (((size_t)(r * HH + h) * DKK) + d1) * DKK;
    float sum = 0.f;
#pragma unroll
    for (int d2 = 0; d2 < DKK; ++d2) sum += M[d2] * Wa[(size_t)((h << 5) + d2) * DD + j];
    wfin_t[(size_t)j * (2 * DD) + kk] = f2bf(0.5f * sum);
}

// ---------------- WMMA GEMM core ----------------
// Wave computes two 16x64 tiles (M-tiles ta,tb) against an LDS-staged B tile.
// LDS rows padded by 8 elems (16B) so 16 lanes x (K+8)*2B stride covers all banks.

template <int K>
__device__ __forceinline__ void gemm_pair(const ushort_t* __restrict__ A,
                                          const ushort_t* Bs,
                                          int ta, int tb, v8f acc0[4], v8f acc1[4]) {
    const int lane = threadIdx.x & 31;
    const int m = lane & 15;
    const int half = lane >> 4;
    const ushort_t* ar0 = A + (size_t)(ta * 16 + m) * K;
    const ushort_t* ar1 = A + (size_t)(tb * 16 + m) * K;
    for (int kk = 0; kk < K; kk += 32) {
        ABFrag a0, a1;
        a0.q[0] = *(const uint4*)(ar0 + kk + 8 * half);
        a0.q[1] = *(const uint4*)(ar0 + kk + 16 + 8 * half);
        a1.q[0] = *(const uint4*)(ar1 + kk + 8 * half);
        a1.q[1] = *(const uint4*)(ar1 + kk + 16 + 8 * half);
#pragma unroll
        for (int s = 0; s < 4; ++s) {
            ABFrag b;
            const ushort_t* bp = Bs + (size_t)(s * 16 + m) * (K + 8) + kk + 8 * half;
            b.q[0] = *(const uint4*)bp;
            b.q[1] = *(const uint4*)(bp + 16);
            acc0[s] = __builtin_amdgcn_wmma_f32_16x16x32_bf16(
                false, a0.v, false, b.v, (short)0, acc0[s], false, false);
            acc1[s] = __builtin_amdgcn_wmma_f32_16x16x32_bf16(
                false, a1.v, false, b.v, (short)0, acc1[s], false, false);
        }
    }
}

template <int K>
__device__ __forceinline__ void stage_B(const ushort_t* __restrict__ Bt, int col64, ushort_t* Bs) {
    const uint4* bsrc = (const uint4*)(Bt + (size_t)col64 * 64 * K);
    for (int i = threadIdx.x; i < 64 * K / 8; i += 128) {
        int row = i / (K / 8);
        int c = i - row * (K / 8);
        ((uint4*)(Bs + row * (K + 8)))[c] = bsrc[i];
    }
}

// k/v/qt projections: z=0 -> k (bf16), z=1 -> v (f32), z=2,3 -> qt[r] (bf16)
__global__ __launch_bounds__(128) void gemm_proj_kernel(
    const ushort_t* __restrict__ xb,
    const ushort_t* __restrict__ wk_t, const ushort_t* __restrict__ wv_t,
    const ushort_t* __restrict__ wq_t,
    const float* __restrict__ bk, const float* __restrict__ bv, const float* __restrict__ bq_c,
    ushort_t* __restrict__ kb, float* __restrict__ vf, ushort_t* __restrict__ qtb) {
    __shared__ ushort_t Bs[64 * (DD + 8)];
    int z = blockIdx.z;
    int col64 = blockIdx.y;
    const ushort_t* Bt;
    const float* bias;
    if (z == 0)      { Bt = wk_t; bias = bk; }
    else if (z == 1) { Bt = wv_t; bias = bv; }
    else             { Bt = wq_t + (size_t)(z - 2) * DD * DD; bias = bq_c + (z - 2) * DD; }
    stage_B<DD>(Bt, col64, Bs);
    __syncthreads();

    int wave = threadIdx.x >> 5, lane = threadIdx.x & 31;
    int nloc = lane & 15, half = lane >> 4;
    for (int pair = blockIdx.x * 4 + wave; pair < NPAIRS; pair += gridDim.x * 4) {
        int ta = 2 * pair;
        int tb = (ta + 1 < MTILES) ? ta + 1 : MTILES - 1;
        v8f acc0[4], acc1[4];
#pragma unroll
        for (int s = 0; s < 4; ++s)
#pragma unroll
            for (int j = 0; j < 8; ++j) { acc0[s][j] = 0.f; acc1[s][j] = 0.f; }
        gemm_pair<DD>(xb, Bs, ta, tb, acc0, acc1);
#pragma unroll
        for (int s = 0; s < 4; ++s) {
            int n = col64 * 64 + s * 16 + nloc;
            float bia = bias[n];
#pragma unroll
            for (int j = 0; j < 8; ++j) {
                int row0 = ta * 16 + j + 8 * half;
                int row1 = tb * 16 + j + 8 * half;
                float v0 = acc0[s][j] + bia;
                float v1 = acc1[s][j] + bia;
                if (z == 1) {
                    vf[(size_t)row0 * DD + n] = v0;
                    vf[(size_t)row1 * DD + n] = v1;
                } else if (z == 0) {
                    kb[(size_t)row0 * DD + n] = f2bf(v0);
                    kb[(size_t)row1 * DD + n] = f2bf(v1);
                } else {
                    qtb[((size_t)(z - 2) * NN + row0) * DD + n] = f2bf(v0);
                    qtb[((size_t)(z - 2) * NN + row1) * DD + n] = f2bf(v1);
                }
            }
        }
    }
}

// final: out = alpha*([vagg0|vagg1] @ Wfin + ba) + (1-alpha)*x
__global__ __launch_bounds__(128) void gemm_final_kernel(
    const ushort_t* __restrict__ vaggb, const ushort_t* __restrict__ wfin_t,
    const float* __restrict__ ba, const float* __restrict__ x,
    const float* __restrict__ skip, float* __restrict__ out) {
    __shared__ ushort_t Bs[64 * (2 * DD + 8)];
    int col64 = blockIdx.y;
    stage_B<2 * DD>(wfin_t, col64, Bs);
    __syncthreads();

    float alpha = 1.0f / (1.0f + __expf(-skip[0]));
    float beta = 1.0f - alpha;
    int wave = threadIdx.x >> 5, lane = threadIdx.x & 31;
    int nloc = lane & 15, half = lane >> 4;
    for (int pair = blockIdx.x * 4 + wave; pair < NPAIRS; pair += gridDim.x * 4) {
        int ta = 2 * pair;
        int tb = (ta + 1 < MTILES) ? ta + 1 : MTILES - 1;
        v8f acc0[4], acc1[4];
#pragma unroll
        for (int s = 0; s < 4; ++s)
#pragma unroll
            for (int j = 0; j < 8; ++j) { acc0[s][j] = 0.f; acc1[s][j] = 0.f; }
        gemm_pair<2 * DD>(vaggb, Bs, ta, tb, acc0, acc1);
#pragma unroll
        for (int s = 0; s < 4; ++s) {
            int n = col64 * 64 + s * 16 + nloc;
            float bia = ba[n];
#pragma unroll
            for (int j = 0; j < 8; ++j) {
                int row0 = ta * 16 + j + 8 * half;
                int row1 = tb * 16 + j + 8 * half;
                out[(size_t)row0 * DD + n] = alpha * (acc0[s][j] + bia) + beta * x[(size_t)row0 * DD + n];
                out[(size_t)row1 * DD + n] = alpha * (acc1[s][j] + bia) + beta * x[(size_t)row1 * DD + n];
            }
        }
    }
}

// ---------------- sparse path: scores + CSR + segment aggregation ----------------

__global__ void edge_att_kernel(const ushort_t* __restrict__ kb, const ushort_t* __restrict__ qtb,
                                const int* __restrict__ src, const int* __restrict__ dst,
                                float* __restrict__ att) {
    int t = blockIdx.x * blockDim.x + threadIdx.x;
    if (t >= RR * EE * HH) return;
    int h = t & 7;
    int ge = t >> 3;               // r*EE + e
    int r = ge / EE;
    int s = src[ge];
    int d = dst[ge];
    const uint4* kp = (const uint4*)(kb + (size_t)s * DD + h * DKK);
    const uint4* qp = (const uint4*)(qtb + ((size_t)r * NN + d) * DD + h * DKK);
    float sum = 0.f;
#pragma unroll
    for (int i = 0; i < 4; ++i) {
        uint4 ka = kp[i], qa = qp[i];
        sum += bflo(ka.x) * bflo(qa.x) + bfhi(ka.x) * bfhi(qa.x);
        sum += bflo(ka.y) * bflo(qa.y) + bfhi(ka.y) * bfhi(qa.y);
        sum += bflo(ka.z) * bflo(qa.z) + bfhi(ka.z) * bfhi(qa.z);
        sum += bflo(ka.w) * bflo(qa.w) + bfhi(ka.w) * bfhi(qa.w);
    }
    att[t] = sum;
}

__global__ void count_kernel(const int* __restrict__ dst, uint_t* __restrict__ cnt) {
    int t = blockIdx.x * blockDim.x + threadIdx.x;
    if (t >= RR * EE) return;
    int r = t / EE;
    atomicAdd(&cnt[r * NN + dst[t]], 1u);
}

// exclusive scan of cnt[RN] -> offs[RN], block partials in bsum
__global__ void scan1_kernel(const uint_t* __restrict__ cnt, uint_t* __restrict__ offs,
                             uint_t* __restrict__ bsum) {
    __shared__ uint_t tot[256];
    int tid = threadIdx.x;
    int base = blockIdx.x * 1024 + tid * 4;
    uint_t v0 = 0, v1 = 0, v2 = 0, v3 = 0;
    if (base + 0 < RN) v0 = cnt[base + 0];
    if (base + 1 < RN) v1 = cnt[base + 1];
    if (base + 2 < RN) v2 = cnt[base + 2];
    if (base + 3 < RN) v3 = cnt[base + 3];
    tot[tid] = v0 + v1 + v2 + v3;
    __syncthreads();
    for (int o = 1; o < 256; o <<= 1) {
        uint_t t = (tid >= o) ? tot[tid - o] : 0u;
        __syncthreads();
        tot[tid] += t;
        __syncthreads();
    }
    if (tid == 255) bsum[blockIdx.x] = tot[255];
    uint_t run = tid ? tot[tid - 1] : 0u;
    if (base + 0 < RN) offs[base + 0] = run; run += v0;
    if (base + 1 < RN) offs[base + 1] = run; run += v1;
    if (base + 2 < RN) offs[base + 2] = run; run += v2;
    if (base + 3 < RN) offs[base + 3] = run;
}

__global__ void scan2_kernel(uint_t* __restrict__ bsum, int nb) {
    if (blockIdx.x == 0 && threadIdx.x == 0) {
        uint_t run = 0;
        for (int i = 0; i < nb; ++i) { uint_t t = bsum[i]; bsum[i] = run; run += t; }
    }
}

__global__ void scan3_kernel(uint_t* __restrict__ offs, const uint_t* __restrict__ bsum,
                             uint_t* __restrict__ cursor) {
    int i = blockIdx.x * blockDim.x + threadIdx.x;
    if (i >= RN) return;
    uint_t o = offs[i] + bsum[i >> 10];
    offs[i] = o;
    cursor[i] = o;
}

__global__ void fill_kernel(const int* __restrict__ dst, uint_t* __restrict__ cursor,
                            int* __restrict__ eidx) {
    int t = blockIdx.x * blockDim.x + threadIdx.x;
    if (t >= RR * EE) return;
    int r = t / EE;
    uint_t pos = atomicAdd(&cursor[r * NN + dst[t]], 1u);
    eidx[pos] = t;
}

// One wave per (relation,dst) segment: softmax + weighted sum, no atomics.
// Lane L handles dims [L*8, L*8+8) (head hh = L>>2); writes vaggb bf16 directly.
__global__ __launch_bounds__(128) void agg_kernel(
    const uint_t* __restrict__ offs, const uint_t* __restrict__ cnt,
    const int* __restrict__ eidx, const float* __restrict__ attb,
    const int* __restrict__ src, const float* __restrict__ vf,
    ushort_t* __restrict__ vaggb) {
    int wave = threadIdx.x >> 5;
    int seg = blockIdx.x * 4 + wave;
    if (seg >= RN) return;
    int lane = threadIdx.x & 31;
    int hh = lane >> 2;
    uint_t beg = offs[seg];
    uint_t n = cnt[seg];
    float acc[8];
#pragma unroll
    for (int j = 0; j < 8; ++j) acc[j] = 0.f;
    if (n > 0) {
        float m = -3.402823466e38f;
        for (uint_t i = 0; i < n; ++i) {
            int ge = eidx[beg + i];
            m = fmaxf(m, attb[(size_t)ge * HH + hh]);
        }
        float den = 0.f;
        for (uint_t i = 0; i < n; ++i) {
            int ge = eidx[beg + i];
            den += __expf(attb[(size_t)ge * HH + hh] - m);
        }
        float rden = 1.f / den;
        for (uint_t i = 0; i < n; ++i) {
            int ge = eidx[beg + i];
            float w = __expf(attb[(size_t)ge * HH + hh] - m) * rden;
            int s = src[ge];
            const float4* vp = (const float4*)(vf + (size_t)s * DD + lane * 8);
            float4 va = vp[0], vb = vp[1];
            acc[0] += w * va.x; acc[1] += w * va.y; acc[2] += w * va.z; acc[3] += w * va.w;
            acc[4] += w * vb.x; acc[5] += w * vb.y; acc[6] += w * vb.z; acc[7] += w * vb.w;
        }
    }
    int r = seg / NN, nn = seg - r * NN;
    uint4 o;
    o.x = (uint_t)f2bf(acc[0]) | ((uint_t)f2bf(acc[1]) << 16);
    o.y = (uint_t)f2bf(acc[2]) | ((uint_t)f2bf(acc[3]) << 16);
    o.z = (uint_t)f2bf(acc[4]) | ((uint_t)f2bf(acc[5]) << 16);
    o.w = (uint_t)f2bf(acc[6]) | ((uint_t)f2bf(acc[7]) << 16);
    *(uint4*)(vaggb + (size_t)nn * (2 * DD) + r * DD + lane * 8) = o;
}

// ---------------- host launcher ----------------

extern "C" void kernel_launch(void* const* d_in, const int* in_sizes, int n_in,
                              void* d_out, int out_size, void* d_ws, size_t ws_size,
                              hipStream_t stream) {
    (void)in_sizes; (void)n_in; (void)out_size; (void)ws_size;
    const float* x       = (const float*)d_in[0];
    const int*   src     = (const int*)d_in[1];
    const int*   dst     = (const int*)d_in[2];
    const float* Wk      = (const float*)d_in[3];
    const float* bk      = (const float*)d_in[4];
    const float* Wq      = (const float*)d_in[5];
    const float* bq      = (const float*)d_in[6];
    const float* Wv      = (const float*)d_in[7];
    const float* bv      = (const float*)d_in[8];
    const float* Wa      = (const float*)d_in[9];
    const float* ba      = (const float*)d_in[10];
    const float* rel_pri = (const float*)d_in[11];
    const float* rel_att = (const float*)d_in[12];
    const float* rel_msg = (const float*)d_in[13];
    const float* skip    = (const float*)d_in[14];
    float* out = (float*)d_out;

    char* ws = (char*)d_ws;
    size_t off = 0;
    auto alloc = [&](size_t bytes) { char* p = ws + off; off = (off + bytes + 255) & ~(size_t)255; return p; };

    // cnt + cursor contiguous (zeroed together each call)
    char*   zzone  = alloc((size_t)2 * RN * 4);           // 800,000 B
    uint_t* cnt    = (uint_t*)zzone;
    uint_t* cursor = (uint_t*)(zzone + (size_t)RN * 4);
    uint_t* offs   = (uint_t*)alloc((size_t)RN * 4);
    uint_t* bsum   = (uint_t*)alloc(1024);
    int*    eidx   = (int*)alloc((size_t)RR * EE * 4);
    float*  attb   = (float*)alloc((size_t)RR * EE * HH * 4);
    ushort_t* xb     = (ushort_t*)alloc((size_t)NN * DD * 2);
    ushort_t* kb     = (ushort_t*)alloc((size_t)NN * DD * 2);
    ushort_t* qtb    = (ushort_t*)alloc((size_t)RR * NN * DD * 2);
    float*    vf     = (float*)alloc((size_t)NN * DD * 4);
    ushort_t* vaggb  = (ushort_t*)alloc((size_t)NN * 2 * DD * 2);
    ushort_t* wk_t   = (ushort_t*)alloc((size_t)DD * DD * 2);
    ushort_t* wv_t   = (ushort_t*)alloc((size_t)DD * DD * 2);
    ushort_t* wq_t   = (ushort_t*)alloc((size_t)RR * DD * DD * 2);
    ushort_t* wfin_t = (ushort_t*)alloc((size_t)2 * DD * DD * 2);
    float*    bq_c   = (float*)alloc((size_t)RR * DD * 4);

    const int nbscan = (RN + 1023) / 1024;              // 98

    zero16_kernel<<<((2 * RN * 4 / 16) + 255) / 256, 256, 0, stream>>>((uint4*)zzone, 2 * RN * 4 / 16);
    conv_x_kernel<<<(NN * DD / 4 + 255) / 256, 256, 0, stream>>>((const float4*)x, xb, NN * DD / 4);
    prep_wkv_kernel<<<(2 * DD * DD + 255) / 256, 256, 0, stream>>>(Wk, Wv, wk_t, wv_t);
    prep_wq_kernel<<<(RR * DD * DD + 255) / 256, 256, 0, stream>>>(Wq, bq, rel_att, rel_pri, wq_t, bq_c);
    prep_wfin_kernel<<<(2 * DD * DD + 255) / 256, 256, 0, stream>>>(Wa, rel_msg, wfin_t);

    gemm_proj_kernel<<<dim3(98, DD / 64, 4), 128, 0, stream>>>(
        xb, wk_t, wv_t, wq_t, bk, bv, bq_c, kb, vf, qtb);

    edge_att_kernel<<<(RR * EE * HH + 255) / 256, 256, 0, stream>>>(kb, qtb, src, dst, attb);

    count_kernel<<<(RR * EE + 255) / 256, 256, 0, stream>>>(dst, cnt);
    scan1_kernel<<<nbscan, 256, 0, stream>>>(cnt, offs, bsum);
    scan2_kernel<<<1, 64, 0, stream>>>(bsum, nbscan);
    scan3_kernel<<<(RN + 255) / 256, 256, 0, stream>>>(offs, bsum, cursor);
    fill_kernel<<<(RR * EE + 255) / 256, 256, 0, stream>>>(dst, cursor, eidx);

    agg_kernel<<<(RN + 3) / 4, 128, 0, stream>>>(offs, cnt, eidx, attb, src, vf, vaggb);

    gemm_final_kernel<<<dim3(98, DD / 64), 128, 0, stream>>>(
        vaggb, wfin_t, ba, x, skip, out);
}
